// E3_TransformerLayer_multi_27754078667286
// MI455X (gfx1250) — matrix operand order repaired and checked by reading, not compiled, not run on hardware
//
#include <hip/hip_runtime.h>
#include <hip/hip_bf16.h>

// Fused equivariant GNN layer for MI455X (gfx1250, wave32).
// Strategy: edge-tiled f16 WMMA (16x16x32, fp32 accum) for the radial MLP
// (8->64->64->64->1024), GEMMs run transposed (rows = hidden/w-channels,
// cols = 16 edges) so each lane owns one edge column and the 1024-wide w
// vector is contracted against P = outer(Ai_src, Ai_dst) straight out of the
// accumulators -- w never touches LDS or HBM. Scatter via fp32 atomics.
// This revision: activations use v_rcp_f32 / v_rsq_f32 instead of IEEE
// division chains (div_scale/refine), cutting per-tile VALU work ~3x.

#define NN 25000     // atoms
#define EE 400000    // edges
#define RMAXF 5.0f

typedef __attribute__((ext_vector_type(16))) _Float16 v16h;
typedef __attribute__((ext_vector_type(2)))  _Float16 v2h;
typedef __attribute__((ext_vector_type(8)))  float    v8f;

__device__ __forceinline__ float silu_f(float x) {
  // x * sigmoid(x) with hardware v_exp_f32 + v_rcp_f32 (no IEEE div chain)
  return x * __builtin_amdgcn_rcpf(1.0f + __expf(-x));
}

__device__ __forceinline__ v8f wmma_fp16(v16h a, v16h b, v8f c) {
  // D = A(16x32,f16) * B(32x16,f16) + C(16x16,f32)
  return __builtin_amdgcn_wmma_f32_16x16x32_f16(false, a, false, b,
                                                (short)0, c, false, false);
}

// Load a 16-half fragment (32 bytes) as v16h via two b128 loads.
__device__ __forceinline__ v16h load_frag(const _Float16* p) {
  union { uint4 u[2]; v16h h; } t;
  const uint4* q = reinterpret_cast<const uint4*>(p);
  t.u[0] = q[0]; t.u[1] = q[1];
  return t.h;
}

// ---------------------------------------------------------------- zero ----
__global__ void zero_kernel(float* __restrict__ out, int* __restrict__ counts) {
  int i = blockIdx.x * blockDim.x + threadIdx.x;
  if (i < NN * 40) out[i] = 0.0f;
  if (i < NN)      counts[i] = 0;
}

// --------------------------------------------------------- weight pack ----
// Pre-pack gw1..gw4 transposed into per-lane WMMA A-fragments (f16).
// A 16x32 f16 layout used here: lane m (0-15): row m, halfs[0..15]=K0..15,
// lane m+16: row m, halfs[0..15]=K16..31 (contiguous k, matching B).
// Fragment table: [0,4)   layer1 (K real = 8, one k-step)
//                 [4,12)  layer2 (4 m-tiles x 2 k-steps)
//                 [12,20) layer3
//                 [20,148) layer4 (64 m-tiles x 2 k-steps)
#define NFRAGS 148
__global__ void pack_kernel(const float* __restrict__ gw1,
                            const float* __restrict__ gw2,
                            const float* __restrict__ gw3,
                            const float* __restrict__ gw4,
                            _Float16* __restrict__ afrag) {
  int gid = blockIdx.x * blockDim.x + threadIdx.x;
  if (gid >= NFRAGS * 32) return;
  int f  = gid >> 5;
  int ln = gid & 31;
  int row16 = ln & 15;
  int hiL   = (ln >> 4) & 1;
  const float* gw; int mt, ks, outDim, realK;
  if (f < 4)       { gw = gw1; mt = f;        ks = 0;          outDim = 64;   realK = 8;  }
  else if (f < 12) { int i = f - 4;  gw = gw2; mt = i >> 1; ks = i & 1; outDim = 64;   realK = 64; }
  else if (f < 20) { int i = f - 12; gw = gw3; mt = i >> 1; ks = i & 1; outDim = 64;   realK = 64; }
  else             { int i = f - 20; gw = gw4; mt = i >> 1; ks = i & 1; outDim = 1024; realK = 64; }
  int row = mt * 16 + row16;
  _Float16* dst = afrag + (size_t)gid * 16;
  for (int hh = 0; hh < 16; ++hh) {
    int klocal = hiL * 16 + hh;
    int K = ks * 32 + klocal;
    float v = (K < realK) ? gw[(size_t)K * outDim + row] : 0.0f;
    dst[hh] = (_Float16)v;
  }
}

// ------------------------------------------------------------ node MLP ----
// Ai[n,8] = (silu(silu(atom_table[A[n]] @ fw1 + fb1) @ fw2 + fb2)) @ fw3 + fb3
__global__ __launch_bounds__(256) void node_kernel(
    const float* __restrict__ atab, const float* __restrict__ fw1,
    const float* __restrict__ fb1,  const float* __restrict__ fw2,
    const float* __restrict__ fb2,  const float* __restrict__ fw3,
    const float* __restrict__ fb3,  const int* __restrict__ A,
    float* __restrict__ AiBuf) {
  __shared__ float W1[16 * 64], W2[64 * 32], W3[32 * 8];
  __shared__ float B1[64], B2[32], B3[8];
  for (int i = threadIdx.x; i < 16 * 64; i += 256) W1[i] = fw1[i];
  for (int i = threadIdx.x; i < 64 * 32; i += 256) W2[i] = fw2[i];
  for (int i = threadIdx.x; i < 32 * 8;  i += 256) W3[i] = fw3[i];
  for (int i = threadIdx.x; i < 64; i += 256) B1[i] = fb1[i];
  for (int i = threadIdx.x; i < 32; i += 256) B2[i] = fb2[i];
  for (int i = threadIdx.x; i < 8;  i += 256) B3[i] = fb3[i];
  __syncthreads();
  int n = blockIdx.x * 256 + threadIdx.x;
  if (n >= NN) return;
  int a = A[n];
  float x[16];
  for (int k = 0; k < 16; ++k) x[k] = atab[a * 16 + k];
  float h1[64];
  for (int j = 0; j < 64; ++j) {
    float acc = B1[j];
    for (int k = 0; k < 16; ++k) acc += x[k] * W1[k * 64 + j];
    h1[j] = silu_f(acc);
  }
  float h2[32];
  for (int j = 0; j < 32; ++j) {
    float acc = B2[j];
    for (int k = 0; k < 64; ++k) acc += h1[k] * W2[k * 32 + j];
    h2[j] = silu_f(acc);
  }
  for (int u = 0; u < 8; ++u) {
    float acc = B3[u];
    for (int k = 0; k < 32; ++k) acc += h2[k] * W3[k * 8 + u];
    AiBuf[(size_t)n * 8 + u] = acc;
  }
}

// ------------------------------------------------------------- counts -----
__global__ void count_kernel(const int* __restrict__ edst, int* __restrict__ counts) {
  int i = blockIdx.x * blockDim.x + threadIdx.x;
  if (i < EE) atomicAdd(&counts[edst[i]], 1);
}

// ----------------------------------------------------- fused edge kernel --
// One wave per 16-edge tile. Transposed GEMMs: D(rows=channels, cols=edges).
// B 32x16 f16 layout assumed: lane n (0-15): col n, halfs = K0..15;
// lane n+16: col n, halfs = K16..31.
__global__ __launch_bounds__(256) void edge_kernel(
    const float* __restrict__ pos,   const float* __restrict__ shifts,
    const float* __restrict__ cellM, const int* __restrict__ batchA,
    const float* __restrict__ gb1,   const float* __restrict__ gb2,
    const float* __restrict__ gb3,   const float* __restrict__ gb4,
    const _Float16* __restrict__ afrag, const float* __restrict__ AiBuf,
    const int* __restrict__ esrc,    const int* __restrict__ edst,
    float* __restrict__ outAcc,      int numTiles) {
  __shared__ __align__(16) _Float16 hbuf[8][16][72]; // per-wave hT repack (pad 72 vs banks)
  __shared__ float Pbuf[8][64][16];                  // per-wave P[p][edge]
  __shared__ float sGb4[1024];
  __shared__ float sGbH[192];                        // gb1|gb2|gb3

  for (int i = threadIdx.x; i < 1024; i += 256) sGb4[i] = gb4[i];
  for (int i = threadIdx.x; i < 192; i += 256)
    sGbH[i] = (i < 64) ? gb1[i] : ((i < 128) ? gb2[i - 64] : gb3[i - 128]);
  __syncthreads();

  const int lane = threadIdx.x & 31;
  const int w    = threadIdx.x >> 5;
  const int e16  = lane & 15;
  const int hi   = lane >> 4;
  const int waveG = blockIdx.x * 8 + w;
  const int waveN = gridDim.x * 8;

  for (int tile = waveG; tile < numTiles; tile += waveN) {
    // ----------------- per-edge setup (lanes 0-15 own one edge each) ------
    float x0[8] = {0, 0, 0, 0, 0, 0, 0, 0};
    float sh[9] = {0, 0, 0, 0, 0, 0, 0, 0, 0};
    int dstIdx = 0;
    int valid  = 0;
    if (hi == 0) {
      int eg = tile * 16 + e16;
      if (eg < EE) {
        valid = 1;
        int srcI = esrc[eg], dstI = edst[eg];
        dstIdx = dstI;
        int b = batchA[srcI];
        const float* C = cellM + (size_t)b * 9;
        float s0 = shifts[(size_t)eg * 3 + 0];
        float s1 = shifts[(size_t)eg * 3 + 1];
        float s2 = shifts[(size_t)eg * 3 + 2];
        float vx = pos[(size_t)dstI * 3 + 0] - pos[(size_t)srcI * 3 + 0] + s0 * C[0] + s1 * C[3] + s2 * C[6];
        float vy = pos[(size_t)dstI * 3 + 1] - pos[(size_t)srcI * 3 + 1] + s0 * C[1] + s1 * C[4] + s2 * C[7];
        float vz = pos[(size_t)dstI * 3 + 2] - pos[(size_t)srcI * 3 + 2] + s0 * C[2] + s1 * C[5] + s2 * C[8];
        float r2 = vx * vx + vy * vy + vz * vz + 1e-12f;
        float inv = __builtin_amdgcn_rsqf(r2);   // v_rsq_f32
        float r   = r2 * inv;                    // r = sqrt(r2)
        float ux = vx * inv, uy = vy * inv, uz = vz * inv;
        const float s3 = 1.7320508075688772f, s15 = 3.872983346207417f;
        const float s5h = 1.118033988749895f, s15h = 1.9364916731037085f;
        sh[0] = 1.0f;
        sh[1] = s3 * ux; sh[2] = s3 * uy; sh[3] = s3 * uz;
        sh[4] = s15 * ux * uy; sh[5] = s15 * uy * uz;
        sh[6] = s5h * (2.0f * uz * uz - ux * ux - uy * uy);
        sh[7] = s15 * ux * uz; sh[8] = s15h * (ux * ux - uy * uy);
        float xr  = fminf(fmaxf(r * (1.0f / RMAXF), 0.0f), 1.0f);
        float cut = (r <= RMAXF) ? 2.8284271247461903f : 0.0f; // sqrt(NB)*cutoff
        #pragma unroll
        for (int k = 0; k < 8; ++k) {
          float t = (xr - (float)k * (1.0f / 7.0f)) * 7.0f;
          x0[k] = __expf(-0.5f * t * t) * cut;
        }
        const float4* Asp = reinterpret_cast<const float4*>(AiBuf + (size_t)srcI * 8);
        const float4* Adp = reinterpret_cast<const float4*>(AiBuf + (size_t)dstI * 8);
        float4 a0 = Asp[0], a1 = Asp[1], d0 = Adp[0], d1 = Adp[1];
        float ai[8] = {a0.x, a0.y, a0.z, a0.w, a1.x, a1.y, a1.z, a1.w};
        float ad[8] = {d0.x, d0.y, d0.z, d0.w, d1.x, d1.y, d1.z, d1.w};
        #pragma unroll
        for (int i = 0; i < 8; ++i)
          #pragma unroll
          for (int j = 0; j < 8; ++j)
            Pbuf[w][i * 8 + j][e16] = ai[i] * ad[j];
      } else {
        #pragma unroll
        for (int p = 0; p < 64; ++p) Pbuf[w][p][e16] = 0.0f;
      }
    }

    // Layer-1 B fragment lives entirely in registers (real K = 8 of 32).
    v16h bL;
    #pragma unroll
    for (int k = 0; k < 16; ++k) bL[k] = (_Float16)0.0f;
    if (hi == 0) {
      #pragma unroll
      for (int k = 0; k < 8; ++k) bL[k] = (_Float16)x0[k];
    }

    _Float16* hrow = &hbuf[w][e16][0];

    // ---------------- layer 1: h1T = silu(gw1T @ x0T + gb1) --------------
    #pragma unroll
    for (int mt = 0; mt < 4; ++mt) {
      v8f c;
      #pragma unroll
      for (int v = 0; v < 8; ++v) c[v] = sGbH[mt * 16 + 8 * hi + v];
      c = wmma_fp16(load_frag(afrag + ((size_t)mt * 512) + lane * 16), bL, c);
      #pragma unroll
      for (int v = 0; v < 8; v += 2) {
        v2h pr; pr[0] = (_Float16)silu_f(c[v]); pr[1] = (_Float16)silu_f(c[v + 1]);
        *reinterpret_cast<v2h*>(hrow + mt * 16 + 8 * hi + v) = pr;
      }
    }
    asm volatile("s_wait_dscnt 0x0" ::: "memory"); // wave-private LDS RAW
    v16h bf0 = load_frag(hrow + hi * 16);          // K 0..31
    v16h bf1 = load_frag(hrow + 32 + hi * 16);     // K 32..63

    // ---------------- layers 2 and 3: 64 -> 64, silu ---------------------
    #pragma unroll
    for (int L = 0; L < 2; ++L) {
      const int fb = 4 + L * 8;
      const float* gb = &sGbH[64 + L * 64];
      v8f acc4[4];
      #pragma unroll
      for (int mt = 0; mt < 4; ++mt) {
        v8f c;
        #pragma unroll
        for (int v = 0; v < 8; ++v) c[v] = gb[mt * 16 + 8 * hi + v];
        c = wmma_fp16(load_frag(afrag + (size_t)(fb + mt * 2 + 0) * 512 + lane * 16), bf0, c);
        c = wmma_fp16(load_frag(afrag + (size_t)(fb + mt * 2 + 1) * 512 + lane * 16), bf1, c);
        acc4[mt] = c;
      }
      #pragma unroll
      for (int mt = 0; mt < 4; ++mt) {
        #pragma unroll
        for (int v = 0; v < 8; v += 2) {
          v2h pr; pr[0] = (_Float16)silu_f(acc4[mt][v]); pr[1] = (_Float16)silu_f(acc4[mt][v + 1]);
          *reinterpret_cast<v2h*>(hrow + mt * 16 + 8 * hi + v) = pr;
        }
      }
      asm volatile("s_wait_dscnt 0x0" ::: "memory");
      bf0 = load_frag(hrow + hi * 16);
      bf1 = load_frag(hrow + 32 + hi * 16);
    }

    // ---- layer 4 (64 -> 1024) fused with tensor-product contraction -----
    // wT tile rows = w columns; each lane holds 8 w values of its edge.
    // q_l[u] = sum_p P[p] * w_l[p,u];  col = 16*mt + 8*hi + v.
    float q0[8] = {0, 0, 0, 0, 0, 0, 0, 0};
    float q1[4] = {0, 0, 0, 0};
    float q2[4] = {0, 0, 0, 0};
    const float* Pe = &Pbuf[w][0][e16];
    #pragma unroll 4
    for (int mt = 0; mt < 32; ++mt) {                 // l=0 block, mul=8: p=col>>3, u=v
      v8f c;
      #pragma unroll
      for (int v = 0; v < 8; ++v) c[v] = sGb4[mt * 16 + 8 * hi + v];
      const _Float16* ab = afrag + (size_t)(20 + mt * 2) * 512 + lane * 16;
      c = wmma_fp16(load_frag(ab), bf0, c);
      c = wmma_fp16(load_frag(ab + 512), bf1, c);
      float Pv = Pe[(2 * mt + hi) * 16];
      #pragma unroll
      for (int v = 0; v < 8; ++v) q0[v] += c[v] * Pv;
    }
    #pragma unroll 4
    for (int mt = 32; mt < 48; ++mt) {                // l=1 block, mul=4
      v8f c;
      #pragma unroll
      for (int v = 0; v < 8; ++v) c[v] = sGb4[mt * 16 + 8 * hi + v];
      const _Float16* ab = afrag + (size_t)(20 + mt * 2) * 512 + lane * 16;
      c = wmma_fp16(load_frag(ab), bf0, c);
      c = wmma_fp16(load_frag(ab + 512), bf1, c);
      int p0 = 4 * (mt - 32) + 2 * hi;
      float Pa = Pe[p0 * 16], Pb = Pe[(p0 + 1) * 16];
      q1[0] += c[0] * Pa + c[4] * Pb;
      q1[1] += c[1] * Pa + c[5] * Pb;
      q1[2] += c[2] * Pa + c[6] * Pb;
      q1[3] += c[3] * Pa + c[7] * Pb;
    }
    #pragma unroll 4
    for (int mt = 48; mt < 64; ++mt) {                // l=2 block, mul=4
      v8f c;
      #pragma unroll
      for (int v = 0; v < 8; ++v) c[v] = sGb4[mt * 16 + 8 * hi + v];
      const _Float16* ab = afrag + (size_t)(20 + mt * 2) * 512 + lane * 16;
      c = wmma_fp16(load_frag(ab), bf0, c);
      c = wmma_fp16(load_frag(ab + 512), bf1, c);
      int p0 = 4 * (mt - 48) + 2 * hi;
      float Pa = Pe[p0 * 16], Pb = Pe[(p0 + 1) * 16];
      q2[0] += c[0] * Pa + c[4] * Pb;
      q2[1] += c[1] * Pa + c[5] * Pb;
      q2[2] += c[2] * Pa + c[6] * Pb;
      q2[3] += c[3] * Pa + c[7] * Pb;
    }

    // combine the two half-wave partials (lane <-> lane^16)
    #pragma unroll
    for (int u = 0; u < 8; ++u) q0[u] += __shfl_xor(q0[u], 16, 32);
    #pragma unroll
    for (int u = 0; u < 4; ++u) q1[u] += __shfl_xor(q1[u], 16, 32);
    #pragma unroll
    for (int u = 0; u < 4; ++u) q2[u] += __shfl_xor(q2[u], 16, 32);

    // ---- scatter: out[dst, f] += alpha * q_l[u] * sh_l[m] ---------------
    if (hi == 0 && valid) {
      float* o = outAcc + (size_t)dstIdx * 40;
      const float alpha = 0.125f;
      #pragma unroll
      for (int u = 0; u < 8; ++u) atomicAdd(o + u, alpha * q0[u] * sh[0]);
      #pragma unroll
      for (int u = 0; u < 4; ++u)
        #pragma unroll
        for (int m = 0; m < 3; ++m)
          atomicAdd(o + 8 + u * 3 + m, alpha * q1[u] * sh[1 + m]);
      #pragma unroll
      for (int u = 0; u < 4; ++u)
        #pragma unroll
        for (int m = 0; m < 5; ++m)
          atomicAdd(o + 20 + u * 5 + m, alpha * q2[u] * sh[4 + m]);
    }
  }
}

// ----------------------------------------------------------- finalize ----
__global__ void final_kernel(float* __restrict__ out, const int* __restrict__ counts) {
  int i = blockIdx.x * blockDim.x + threadIdx.x;
  if (i < NN * 40) {
    int c = counts[i / 40];
    if (c < 1) c = 1;
    out[i] = out[i] / (float)c;
  }
}

// ------------------------------------------------------------- launch ----
extern "C" void kernel_launch(void* const* d_in, const int* in_sizes, int n_in,
                              void* d_out, int out_size, void* d_ws, size_t ws_size,
                              hipStream_t stream) {
  const float* pos    = (const float*)d_in[0];
  const float* shifts = (const float*)d_in[1];
  const float* cellM  = (const float*)d_in[2];
  const float* atab   = (const float*)d_in[3];
  const float* fw1 = (const float*)d_in[4];
  const float* fb1 = (const float*)d_in[5];
  const float* fw2 = (const float*)d_in[6];
  const float* fb2 = (const float*)d_in[7];
  const float* fw3 = (const float*)d_in[8];
  const float* fb3 = (const float*)d_in[9];
  const float* gw1 = (const float*)d_in[10];
  const float* gb1 = (const float*)d_in[11];
  const float* gw2 = (const float*)d_in[12];
  const float* gb2 = (const float*)d_in[13];
  const float* gw3 = (const float*)d_in[14];
  const float* gb3 = (const float*)d_in[15];
  const float* gw4 = (const float*)d_in[16];
  const float* gb4 = (const float*)d_in[17];
  const int* A      = (const int*)d_in[18];
  const int* batchA = (const int*)d_in[19];
  const int* esrc   = (const int*)d_in[20];
  const int* edst   = (const int*)d_in[21];
  float* out = (float*)d_out;

  // workspace layout (~1.06 MB): counts | AiBuf | packed A-fragments
  char* ws = (char*)d_ws;
  int*       counts = (int*)ws;                     // NN * 4      bytes
  float*     AiBuf  = (float*)(ws + 102400);        // NN * 8 * 4  bytes
  _Float16*  afrag  = (_Float16*)(ws + 902400);     // 148*512*2   bytes
  (void)in_sizes; (void)n_in; (void)out_size; (void)ws_size;

  zero_kernel<<<(NN * 40 + 255) / 256, 256, 0, stream>>>(out, counts);
  pack_kernel<<<(NFRAGS * 32 + 255) / 256, 256, 0, stream>>>(gw1, gw2, gw3, gw4, afrag);
  node_kernel<<<(NN + 255) / 256, 256, 0, stream>>>(atab, fw1, fb1, fw2, fb2, fw3, fb3, A, AiBuf);
  count_kernel<<<(EE + 255) / 256, 256, 0, stream>>>(edst, counts);

  int tiles  = (EE + 15) / 16;          // 25000 tiles of 16 edges
  int blocks = (tiles + 7) / 8;         // 8 waves per 256-thread block
  edge_kernel<<<blocks, 256, 0, stream>>>(pos, shifts, cellM, batchA,
                                          gb1, gb2, gb3, gb4, afrag, AiBuf,
                                          esrc, edst, out, tiles);

  final_kernel<<<(NN * 40 + 255) / 256, 256, 0, stream>>>(out, counts);
}
